// SpikingActorSAC_21912923144774
// MI455X (gfx1250) — matrix operand order repaired and checked by reading, not compiled, not run on hardware
//
#include <hip/hip_runtime.h>
#include <math.h>

typedef unsigned short u16;
typedef unsigned int   u32;
typedef __attribute__((ext_vector_type(16))) __bf16   v16bf;
typedef __attribute__((ext_vector_type(8)))  __bf16   v8bf;   // bf16 C/D tile (4 VGPRs)
typedef __attribute__((ext_vector_type(8)))  float    v8f;
typedef __attribute__((ext_vector_type(4)))  u32      u32x4;
typedef __attribute__((ext_vector_type(4)))  float    f32x4;
typedef __attribute__((ext_vector_type(2)))  _Float16 h2f;    // packed f16 pair (1 VGPR)

#define B_DIM 8192
#define N_DIM 128
#define H_DIM 1024
#define A_DIM 32
#define O_DIM 64
#define T_STEPS 8
#define TILE_M 64          // batch rows per workgroup
#define NTHREADS 512       // 16 wave32; each wave owns 64 H-columns

union Frag16 {
  v16bf v;
  u16   h[16];
  u32x4 q[2];
};

__device__ __forceinline__ u16 f2bf(float f) {
  u32 u = __builtin_bit_cast(u32, f);
  u += 0x7FFFu + ((u >> 16) & 1u);   // round-to-nearest-even
  return (u16)(u >> 16);
}
__device__ __forceinline__ float bf2f(u16 h) {
  u32 u = ((u32)h) << 16;
  return __builtin_bit_cast(float, u);
}

// A-matrix bf16 16x32 fragment per ISA layout:
//   lanes 0-15 : row M=lane,    K = k0+[0..7] then k0+[16..23]
//   lanes 16-31: row M=lane-16, K = k0+[8..15] then k0+[24..31]
__device__ __forceinline__ v16bf load_fragA(const u16* base, int stride, int rbase,
                                            int k0, int lane) {
  int r = lane & 15, half = lane >> 4;
  const u16* p = base + (size_t)(rbase + r) * stride + k0 + half * 8;
  Frag16 u;
  u.q[0] = *(const u32x4*)(p);
  u.q[1] = *(const u32x4*)(p + 16);
  return u.v;
}

// B-matrix bf16 32x16 fragment (column-per-lane, same K order as A):
//   lanes 0-15 : col N=lane,    K = k0+[0..7] then k0+[16..23]
//   lanes 16-31: col N=lane-16, K = k0+[8..15] then k0+[24..31]
__device__ __forceinline__ v16bf load_fragB(const u16* base, int stride, int nbase,
                                            int k0, int lane) {
  int c = lane & 15, half = lane >> 4;
  const u16* p = base + (size_t)(nbase + c) * stride + k0 + half * 8;
  Frag16 u;
  u.q[0] = *(const u32x4*)(p);
  u.q[1] = *(const u32x4*)(p + 16);
  return u.v;
}

// A fragment straight from f32 global (state matrix), converted to bf16
__device__ __forceinline__ v16bf load_fragA_f32(const float* base, int stride, int rbase,
                                                int k0, int lane) {
  int r = lane & 15, half = lane >> 4;
  const float* p = base + (size_t)(rbase + r) * stride + k0 + half * 8;
  float f[16];
  *(f32x4*)&f[0]  = *(const f32x4*)(p);
  *(f32x4*)&f[4]  = *(const f32x4*)(p + 4);
  *(f32x4*)&f[8]  = *(const f32x4*)(p + 16);
  *(f32x4*)&f[12] = *(const f32x4*)(p + 20);
  Frag16 u;
#pragma unroll
  for (int i = 0; i < 16; ++i) u.h[i] = f2bf(f[i]);
  return u.v;
}

__device__ __forceinline__ v8f wmma_bf16(v16bf a, v16bf b, v8f c) {
  return __builtin_amdgcn_wmma_f32_16x16x32_bf16(false, a, false, b, (short)0, c,
                                                 false, false);
}
// bf16 C/D variant: halves the persistent accumulator footprint (4 VGPRs/tile)
__device__ __forceinline__ v8bf wmma_bf16c(v16bf a, v16bf b, v8bf c) {
  return __builtin_amdgcn_wmma_bf16_16x16x32_bf16(false, a, false, b, (short)0, c,
                                                  false, false);
}

// ---------------------------------------------------------------------------
// Prep: convert + transpose weights to bf16, N-major, in workspace.
//   w1T [1024][128], w2T [1024][1024], w3T [64][1024]
// ---------------------------------------------------------------------------
__global__ void cvt_weights_kernel(const float* __restrict__ w1,
                                   const float* __restrict__ w2,
                                   const float* __restrict__ w3,
                                   u16* __restrict__ w1T,
                                   u16* __restrict__ w2T,
                                   u16* __restrict__ w3T) {
  int i = blockIdx.x * blockDim.x + threadIdx.x;
  const int S1 = H_DIM * N_DIM;          // 131072
  const int S2 = H_DIM * H_DIM;          // 1048576
  if (i < S1) {
    int n = i / N_DIM, k = i % N_DIM;    // w1: [K=128][N=1024]
    w1T[i] = f2bf(w1[(size_t)k * H_DIM + n]);
  } else if (i < S1 + S2) {
    int j = i - S1;
    int n = j >> 10, k = j & 1023;       // w2: [K=1024][N=1024]
    w2T[j] = f2bf(w2[(size_t)k * H_DIM + n]);
  } else if (i < S1 + S2 + O_DIM * H_DIM) {
    int j = i - S1 - S2;
    int n = j >> 10, k = j & 1023;       // w3: [K=1024][N=64]
    w3T[j] = f2bf(w3[(size_t)k * O_DIM + n]);
  }
}

// IF-1 membrane update, macro-expanded: every v1p index is a literal constant.
#define IF1_PAIR(J, P)                                                       \
    {                                                                        \
      h2f vv = v1p[(J) * 4 + (P)];                                           \
      float h0 = (float)vv[0] + bf2f(xu.h[2 * (P)]);                         \
      float h1 = (float)vv[1] + bf2f(xu.h[2 * (P) + 1]);                     \
      bool s0 = (h0 - 1.0f) >= 0.0f;                                         \
      bool s1 = (h1 - 1.0f) >= 0.0f;                                         \
      vv[0] = (_Float16)(s0 ? 0.0f : h0);                                    \
      vv[1] = (_Float16)(s1 ? 0.0f : h1);                                    \
      v1p[(J) * 4 + (P)] = vv;                                               \
      su.h[2 * (P)]     = s0 ? (u16)0x3F80 : (u16)0;                         \
      su.h[2 * (P) + 1] = s1 ? (u16)0x3F80 : (u16)0;                         \
    }

#define IF1_J(J)                                                             \
    {                                                                        \
      const int idx = (J) * (NTHREADS * 8) + tid * 8;                        \
      union { u32x4 q; u16 h[8]; } xu, su;                                   \
      xu.q = *(const u32x4*)&x1s[idx];                                       \
      IF1_PAIR(J, 0) IF1_PAIR(J, 1) IF1_PAIR(J, 2) IF1_PAIR(J, 3)            \
      *(u32x4*)&ss[idx] = su.q;                                              \
    }

#define V1Z(I)  v1p[I] = z2;
#define V1Z8(I) V1Z(I) V1Z(I+1) V1Z(I+2) V1Z(I+3) V1Z(I+4) V1Z(I+5) V1Z(I+6) V1Z(I+7)

// ---------------------------------------------------------------------------
// Fused SNN actor: 64 batch rows per workgroup, whole T=8 recurrence on-chip.
//   v1 : packed f16 pairs in VGPRs (64/thread)
//   v2 : bf16, lives in bf16 WMMA C/D accumulators (64/thread)
//   x1 : bf16 in LDS (computed once; state@w1 is time-invariant)
//   s1/s2 : bf16 spikes in one shared LDS buffer (barrier-phased)
// ---------------------------------------------------------------------------
__global__ __launch_bounds__(NTHREADS) void snn_actor_kernel(
    const float* __restrict__ state, const float* __restrict__ b1,
    const float* __restrict__ b2, const float* __restrict__ b3,
    const float* __restrict__ eps, const u16* __restrict__ w1T,
    const u16* __restrict__ w2T, const u16* __restrict__ w3T,
    float* __restrict__ out) {
  __shared__ __align__(16) u16 x1s[TILE_M * H_DIM];  // 128 KB bf16 x1 (const over t)
  __shared__ __align__(16) u16 ss [TILE_M * H_DIM];  // 128 KB spikes s1 then s2

  const int tid  = threadIdx.x;
  const int lane = tid & 31;
  const int wave = tid >> 5;          // 16 waves
  const int row0 = blockIdx.x * TILE_M;
  const int n0   = wave * 64;         // each wave owns 64 H-columns

  // ---- GEMM1 (once): x1 = state @ w1 + b1 -> x1s (bf16); fully unrolled so
  // the time loop's GEMM2 becomes the first inner loop in the emitted asm ----
  {
    v16bf af[4][4];
#pragma unroll
    for (int mb = 0; mb < 4; ++mb)
#pragma unroll
      for (int kb = 0; kb < 4; ++kb)
        af[mb][kb] = load_fragA_f32(state, N_DIM, row0 + mb * 16, kb * 32, lane);

#pragma unroll
    for (int nb = 0; nb < 4; ++nb) {
      v8f c[4];
#pragma unroll
      for (int mb = 0; mb < 4; ++mb)
        c[mb] = (v8f){0.f, 0.f, 0.f, 0.f, 0.f, 0.f, 0.f, 0.f};
#pragma unroll
      for (int kb = 0; kb < 4; ++kb) {
        v16bf bf = load_fragB(w1T, N_DIM, n0 + nb * 16, kb * 32, lane);
#pragma unroll
        for (int mb = 0; mb < 4; ++mb) c[mb] = wmma_bf16(af[mb][kb], bf, c[mb]);
      }
      int col = n0 + nb * 16 + (lane & 15);
      float bv = b1[col];
      int rb = (lane >> 4) * 8;
#pragma unroll
      for (int mb = 0; mb < 4; ++mb)
#pragma unroll
        for (int r = 0; r < 8; ++r)
          x1s[(mb * 16 + rb + r) * H_DIM + col] = f2bf(c[mb][r] + bv);
    }
  }

  // ---- persistent register state ----
  h2f v1p[64];                        // f16 membrane v1: 64x1024 over 512 threads
  {
    const h2f z2 = (h2f){(_Float16)0.f, (_Float16)0.f};
    V1Z8(0) V1Z8(8) V1Z8(16) V1Z8(24) V1Z8(32) V1Z8(40) V1Z8(48) V1Z8(56)
  }

  v8bf acc2[4][4];                    // v2 lives in bf16 WMMA C/D registers
#pragma unroll
  for (int mb = 0; mb < 4; ++mb)
#pragma unroll
    for (int nb = 0; nb < 4; ++nb)
#pragma unroll
      for (int r = 0; r < 8; ++r)
        acc2[mb][nb][r] = (__bf16)0.f;
  v8f v3a = {0.f, 0.f, 0.f, 0.f, 0.f, 0.f, 0.f, 0.f};

  const int mb3 = wave & 3, nb3 = wave >> 2;   // wave's 16x16 tile of 64x64 out3
  const float b3v = b3[nb3 * 16 + (lane & 15)];

  __syncthreads();

  // =============================== time loop ===============================
#pragma unroll 1
  for (int t = 0; t < T_STEPS; ++t) {
    // --- IF node 1 (x1 from LDS, v1 in f16-pair regs), macro-expanded ---
    {
      IF1_J(0)  IF1_J(1)  IF1_J(2)  IF1_J(3)
      IF1_J(4)  IF1_J(5)  IF1_J(6)  IF1_J(7)
      IF1_J(8)  IF1_J(9)  IF1_J(10) IF1_J(11)
      IF1_J(12) IF1_J(13) IF1_J(14) IF1_J(15)
    }
    __syncthreads();   // s1 visible

    // --- GEMM2: acc2(v2, bf16) += s1 @ w2   (M=64, N=64/wave, K=1024) ---
#pragma unroll 1
    for (int kb = 0; kb < 32; ++kb) {
      int k0 = kb * 32;
      v16bf a[4];
#pragma unroll
      for (int mb = 0; mb < 4; ++mb)
        a[mb] = load_fragA((const u16*)ss, H_DIM, mb * 16, k0, lane);
#pragma unroll
      for (int nb = 0; nb < 4; ++nb) {
        v16bf bf = load_fragB(w2T, H_DIM, n0 + nb * 16, k0, lane);
#pragma unroll
        for (int mb = 0; mb < 4; ++mb)
          acc2[mb][nb] = wmma_bf16c(a[mb], bf, acc2[mb][nb]);
      }
    }
    __syncthreads();   // everyone done reading s1 before s2 overwrites ss

    // --- IF node 2 epilogue: h2=acc2+b2; s2 spikes -> ss; acc2 <- v2' ---
    {
      int rb = (lane >> 4) * 8;
#pragma unroll
      for (int mb = 0; mb < 4; ++mb)
#pragma unroll
        for (int nb = 0; nb < 4; ++nb) {
          int col = n0 + nb * 16 + (lane & 15);
          float bv = b2[col];
#pragma unroll
          for (int r = 0; r < 8; ++r) {
            float h2 = (float)acc2[mb][nb][r] + bv;
            bool sp = (h2 - 1.0f) >= 0.0f;
            acc2[mb][nb][r] = (__bf16)(sp ? 0.0f : h2);
            ss[(mb * 16 + rb + r) * H_DIM + col] = sp ? (u16)0x3F80 : (u16)0;
          }
        }
    }
    __syncthreads();   // s2 visible

    // --- GEMM3 + LIF: x3 = s2 @ w3 + b3; v3 = 0.5*(v3 + x3) ---
    {
      v8f c = {0.f, 0.f, 0.f, 0.f, 0.f, 0.f, 0.f, 0.f};
#pragma unroll 1
      for (int kb = 0; kb < 32; ++kb) {
        int k0 = kb * 32;
        v16bf a  = load_fragA((const u16*)ss, H_DIM, mb3 * 16, k0, lane);
        v16bf bf = load_fragB(w3T, H_DIM, nb3 * 16, k0, lane);
        c = wmma_bf16(a, bf, c);
      }
#pragma unroll
      for (int r = 0; r < 8; ++r)
        v3a[r] = 0.5f * (v3a[r] + c[r] + b3v);
    }
    __syncthreads();   // done reading s2 before next step rewrites ss
  }

  // ---- tanh-Gaussian head: park v3 in LDS (reusing x1s), then per-row ----
  {
    float* v3s = (float*)x1s;
    int col = nb3 * 16 + (lane & 15);
    int rb = mb3 * 16 + (lane >> 4) * 8;
#pragma unroll
    for (int r = 0; r < 8; ++r) v3s[(rb + r) * O_DIM + col] = v3a[r];
  }
  __syncthreads();

  if (tid < TILE_M) {
    const float* v3s = (const float*)x1s;
    int grow = row0 + tid;
    float lp = 0.0f;
#pragma unroll 1
    for (int a = 0; a < A_DIM; ++a) {
      float mu = v3s[tid * O_DIM + a];
      float ls = v3s[tid * O_DIM + A_DIM + a];
      ls = fminf(fmaxf(ls, -20.0f), 2.0f);
      float sd = __expf(ls);                  // hardware exp, always inline
      float e  = eps[(size_t)grow * A_DIM + a];
      float z  = mu + sd * e;
      // tanh via hardware exp: tanh(z) = sign(z) * (1 - e^{-2|z|}) / (1 + e^{-2|z|})
      float az = fabsf(z);
      float tt = __expf(-2.0f * az);
      float th = (1.0f - tt) / (1.0f + tt);
      float act = (z < 0.0f) ? -th : th;
      out[(size_t)grow * A_DIM + a] = act;
      lp += -0.5f * e * e - ls - 0.9189385332046727f
            - __logf(1.0f - act * act + 1e-7f);
    }
    out[(size_t)B_DIM * A_DIM + grow] = lp;
  }
}

// ---------------------------------------------------------------------------
extern "C" void kernel_launch(void* const* d_in, const int* in_sizes, int n_in,
                              void* d_out, int out_size, void* d_ws, size_t ws_size,
                              hipStream_t stream) {
  (void)in_sizes; (void)n_in; (void)out_size; (void)ws_size;
  const float* state = (const float*)d_in[0];
  const float* w1    = (const float*)d_in[1];
  const float* b1    = (const float*)d_in[2];
  const float* w2    = (const float*)d_in[3];
  const float* b2    = (const float*)d_in[4];
  const float* w3    = (const float*)d_in[5];
  const float* b3    = (const float*)d_in[6];
  const float* eps   = (const float*)d_in[7];
  float* out = (float*)d_out;

  // workspace: bf16 N-major weights (~2.4 MB)
  u16* w1T = (u16*)d_ws;                       // 1024*128
  u16* w2T = w1T + (size_t)H_DIM * N_DIM;      // 1024*1024
  u16* w3T = w2T + (size_t)H_DIM * H_DIM;      // 64*1024

  const int total = H_DIM * N_DIM + H_DIM * H_DIM + O_DIM * H_DIM;
  cvt_weights_kernel<<<(total + 255) / 256, 256, 0, stream>>>(w1, w2, w3, w1T, w2T, w3T);
  snn_actor_kernel<<<B_DIM / TILE_M, NTHREADS, 0, stream>>>(state, b1, b2, b3, eps,
                                                            w1T, w2T, w3T, out);
}
